// SwinBlock_31155692765619
// MI455X (gfx1250) — matrix operand order repaired and checked by reading, not compiled, not run on hardware
//
#include <hip/hip_runtime.h>
#include <cstdint>

typedef __bf16 bf16;
typedef __bf16 bf16x16 __attribute__((ext_vector_type(16)));
typedef float  floatx8 __attribute__((ext_vector_type(8)));
typedef int    v4i_vs  __attribute__((vector_size(16)));

#define DIMC     384
#define HEADS    12
#define WINSZ    7
#define NTOK     49      // tokens per window
#define SHIFT3   3
#define MROWS    100352  // 2048 windows * 49 tokens
#define BWIN     2048
#define QKSCALE  0.17677669529663687f  // 32^-0.5

// ---- CDNA5 async global->LDS copy (GLOBAL_LOAD_ASYNC_TO_LDS_B128), guarded --
#if defined(__gfx1250__) && __has_builtin(__builtin_amdgcn_global_load_async_to_lds_b128) && __has_builtin(__builtin_amdgcn_s_wait_asynccnt)
#define USE_ASYNC 1
#else
#define USE_ASYNC 0
#endif

static __device__ __forceinline__ void cp16(const void* g, void* l) {
#if USE_ASYNC
  __builtin_amdgcn_global_load_async_to_lds_b128(
      (__attribute__((address_space(1))) v4i_vs*)(g),
      (__attribute__((address_space(3))) v4i_vs*)(l), 0, 0);
#else
  *(uint4*)l = *(const uint4*)g;
#endif
}

// ---------------------------------------------------------------------------
// WMMA fragment loaders from LDS (per CDNA5 ISA 16-bit layouts, wave32)
// A 16x32: lanes0-15 M=lane, K pairs {0..7}+{16..23}; lanes16-31 K base +8.
//   -> two 16B chunks at +kb and +kb+8 dwords.
// B 32x16: lane = column; same K order as A (lanes0-15 K {0..7}+{16..23}, lanes16-31 +8).
//   -> two 16B chunks at +kb and +kb+8 dwords.
// ---------------------------------------------------------------------------
static __device__ __forceinline__ bf16x16
fragA(const uint32_t* lds, int row, int strideDw, int kkDw, int lane) {
  const uint32_t* p = lds + row * strideDw + kkDw + ((lane & 16) ? 4 : 0);
  union { bf16x16 v; uint4 q[2]; } f;
  f.q[0] = *(const uint4*)(p);
  f.q[1] = *(const uint4*)(p + 8);
  return f.v;
}

static __device__ __forceinline__ bf16x16
fragB(const uint32_t* lds, int row, int strideDw, int kkDw, int lane) {
  const uint32_t* p = lds + row * strideDw + kkDw + ((lane & 16) ? 4 : 0);
  union { bf16x16 v; uint4 q[2]; } f;
  f.q[0] = *(const uint4*)(p);
  f.q[1] = *(const uint4*)(p + 8);
  return f.v;
}

static __device__ __forceinline__ floatx8
wmma_bf16(bf16x16 a, bf16x16 b, floatx8 c) {
  return __builtin_amdgcn_wmma_f32_16x16x32_bf16(false, a, false, b,
                                                 (short)0, c, false, false);
}

// ---------------------------------------------------------------------------
// Weight convert + transpose: W (K x N, f32 row-major) -> Wt (N x K, bf16)
// ---------------------------------------------------------------------------
__global__ __launch_bounds__(256) void k_wconv(const float* __restrict__ W,
                                               bf16* __restrict__ Wt,
                                               int K, int N) {
  int idx = blockIdx.x * 256 + threadIdx.x;
  if (idx < K * N) {
    int k = idx / N, n = idx - k * N;
    Wt[(size_t)n * K + k] = (bf16)W[idx];
  }
}

// ---------------------------------------------------------------------------
// LayerNorm, wave-per-token.  SHIFTED: gather through roll(-3,-3) + window
// partition (window-token order out); else identity row mapping.
// ---------------------------------------------------------------------------
template <bool SHIFTED>
__global__ __launch_bounds__(256) void k_ln(const float* __restrict__ x,
                                            const float* __restrict__ gw,
                                            const float* __restrict__ gb,
                                            bf16* __restrict__ out) {
  const int wave = threadIdx.x >> 5;
  const int lane = threadIdx.x & 31;
  const int tok = blockIdx.x * 8 + wave;  // window-token order index
  const float* src;
  if (SHIFTED) {
    int bimg = tok / 3136;
    int rem  = tok - bimg * 3136;
    int wg   = rem / NTOK;
    int t    = rem - wg * NTOK;
    int r = t / WINSZ, c = t - r * WINSZ;
    int h = (wg >> 3) * WINSZ + r + SHIFT3; if (h >= 56) h -= 56;
    int w = (wg & 7)  * WINSZ + c + SHIFT3; if (w >= 56) w -= 56;
    src = x + ((size_t)bimg * 3136 + h * 56 + w) * DIMC;
  } else {
    src = x + (size_t)tok * DIMC;
  }
  float v[12];
  float s = 0.f, s2 = 0.f;
  #pragma unroll
  for (int i = 0; i < 12; ++i) {
    v[i] = src[lane + 32 * i];
    s += v[i];
    s2 += v[i] * v[i];
  }
  #pragma unroll
  for (int off = 16; off > 0; off >>= 1) {
    s  += __shfl_xor(s,  off, 32);
    s2 += __shfl_xor(s2, off, 32);
  }
  const float mu  = s * (1.f / DIMC);
  const float var = s2 * (1.f / DIMC) - mu * mu;
  const float rs  = rsqrtf(var + 1e-5f);
  bf16* dst = out + (size_t)tok * DIMC;
  #pragma unroll
  for (int i = 0; i < 12; ++i) {
    int ch = lane + 32 * i;
    dst[ch] = (bf16)((v[i] - mu) * rs * gw[ch] + gb[ch]);
  }
}

// ---------------------------------------------------------------------------
// Tiled bf16 WMMA GEMM: C(MxN) = A(MxK) * Bt(NxK)^T + bias, fused epilogues.
// Block tile 128x64, BK=32, double-buffered LDS, software-pipelined staging
// (async global->LDS on CDNA5). 8 waves, each owns a 2x2 grid of 16x16
// tiles: 4 WMMAs per K-step from 2 A + 2 B fragments.
// ---------------------------------------------------------------------------
enum { MODE_QKV = 0, MODE_GELU = 1, MODE_PROJ = 2, MODE_MLP2 = 3 };

template <int MODE>
__global__ __launch_bounds__(256) void k_gemm(const bf16* __restrict__ A,
                                              const bf16* __restrict__ Bt,
                                              const float* __restrict__ bias,
                                              void* __restrict__ outv,
                                              const float* __restrict__ res,
                                              int N, int K) {
  __shared__ __align__(16) bf16 As[2][128 * 32];
  __shared__ __align__(16) bf16 Bs[2][64 * 32];
  const int lane = threadIdx.x & 31;
  const int wave = threadIdx.x >> 5;
  const int m0 = blockIdx.y * 128;
  const int n0 = blockIdx.x * 64;
  const int wr = wave >> 1;  // 0..3: m tiles {2wr, 2wr+1}
  const int wc = wave & 1;   // 0..1: n tiles {2wc, 2wc+1}
  const floatx8 zero = {};
  floatx8 acc[2][2] = {{zero, zero}, {zero, zero}};

  const int ldrow = threadIdx.x >> 2;       // 0..63
  const int ldch  = (threadIdx.x & 3) * 8;  // bf16 offset within 32-wide row
  const bf16* gA0 = A  + (size_t)(m0 + ldrow) * K + ldch;
  const bf16* gA1 = A  + (size_t)(m0 + 64 + ldrow) * K + ldch;
  const bf16* gB  = Bt + (size_t)(n0 + ldrow) * K + ldch;
  bf16* lA0 = &As[0][ldrow * 32 + ldch];
  bf16* lA1 = &As[0][(ldrow + 64) * 32 + ldch];
  bf16* lB  = &Bs[0][ldrow * 32 + ldch];
  const int bufA = 128 * 32;  // bf16 elements per A buffer
  const int bufB = 64 * 32;

  const int nIter = K >> 5;
  // prologue stage -> buffer 0
  cp16(gA0, lA0);
  cp16(gA1, lA1);
  cp16(gB, lB);

  for (int it = 0; it < nIter; ++it) {
#if USE_ASYNC
    __builtin_amdgcn_s_wait_asynccnt(0);
#endif
    __syncthreads();
    if (it + 1 < nIter) {
      const int nb = (it + 1) & 1;
      const int k0 = (it + 1) * 32;
      cp16(gA0 + k0, lA0 + nb * bufA);
      cp16(gA1 + k0, lA1 + nb * bufA);
      cp16(gB + k0, lB + nb * bufB);
    }
    const uint32_t* a32 = (const uint32_t*)As[it & 1];
    const uint32_t* b32 = (const uint32_t*)Bs[it & 1];
    bf16x16 a0 = fragA(a32, wr * 32 + (lane & 15), 16, 0, lane);
    bf16x16 a1 = fragA(a32, wr * 32 + 16 + (lane & 15), 16, 0, lane);
    bf16x16 b0 = fragB(b32, wc * 32 + (lane & 15), 16, 0, lane);
    bf16x16 b1 = fragB(b32, wc * 32 + 16 + (lane & 15), 16, 0, lane);
    acc[0][0] = wmma_bf16(a0, b0, acc[0][0]);
    acc[0][1] = wmma_bf16(a0, b1, acc[0][1]);
    acc[1][0] = wmma_bf16(a1, b0, acc[1][0]);
    acc[1][1] = wmma_bf16(a1, b1, acc[1][1]);
  }

  const int rowHi = (lane & 16) ? 8 : 0;
  const int colB = n0 + wc * 32 + (lane & 15);
  const float bz0 = bias[colB];
  const float bz1 = bias[colB + 16];
  #pragma unroll
  for (int mt = 0; mt < 2; ++mt) {
    #pragma unroll
    for (int j = 0; j < 8; ++j) {
      const int row = m0 + (wr * 2 + mt) * 16 + rowHi + j;
      float v0 = acc[mt][0][j] + bz0;
      float v1 = acc[mt][1][j] + bz1;
      if (MODE == MODE_QKV) {
        if (colB < DIMC) v0 *= QKSCALE;
        if (colB + 16 < DIMC) v1 *= QKSCALE;
        bf16* o = (bf16*)outv + (size_t)row * N + colB;
        o[0] = (bf16)v0;
        o[16] = (bf16)v1;
      } else if (MODE == MODE_GELU) {
        v0 = 0.5f * v0 * (1.f + erff(v0 * 0.70710678118654752f));
        v1 = 0.5f * v1 * (1.f + erff(v1 * 0.70710678118654752f));
        bf16* o = (bf16*)outv + (size_t)row * N + colB;
        o[0] = (bf16)v0;
        o[16] = (bf16)v1;
      } else if (MODE == MODE_PROJ) {
        // window-reverse + roll(+3,+3) scatter, add residual x
        int wg = row / NTOK;
        int tt = row - wg * NTOK;
        int bimg = wg >> 6, wloc = wg & 63;
        int r = tt / WINSZ, c = tt - r * WINSZ;
        int h = (wloc >> 3) * WINSZ + r + SHIFT3; if (h >= 56) h -= 56;
        int w = (wloc & 7)  * WINSZ + c + SHIFT3; if (w >= 56) w -= 56;
        size_t dst = ((size_t)bimg * 3136 + h * 56 + w) * (size_t)DIMC + colB;
        ((float*)outv)[dst] = res[dst] + v0;
        ((float*)outv)[dst + 16] = res[dst + 16] + v1;
      } else {  // MODE_MLP2: plain residual add, fp32 out
        size_t dst = (size_t)row * N + colB;
        ((float*)outv)[dst] = res[dst] + v0;
        ((float*)outv)[dst + 16] = res[dst + 16] + v1;
      }
    }
  }
}

// ---------------------------------------------------------------------------
// Windowed attention, one block per (window, head). 49 tokens padded to 64.
// S = Q*K^T + addT (precomputed rel-bias + shift-mask table), softmax,
// O = P*V.  All matmuls WMMA; epilogues fully branchless.
// ---------------------------------------------------------------------------
static __device__ __forceinline__ int rgn(int y) {
  return y < 49 ? 0 : (y < 53 ? 1 : 2);
}

__global__ __launch_bounds__(256) void k_attn(const bf16* __restrict__ qkv,
                                              const float* __restrict__ relb,
                                              bf16* __restrict__ outp) {
  __shared__ __align__(16) bf16  Qs[64 * 32];
  __shared__ __align__(16) bf16  Ks[64 * 32];
  __shared__ __align__(16) bf16  Vt[32 * 64];   // transposed: [d][m]
  __shared__ __align__(16) float Ss[64 * 64];
  __shared__ __align__(16) bf16  Ps[64 * 64];
  __shared__ __align__(16) float addT[64 * 64]; // rel-bias + mask (+ -1e30 pad)
  __shared__ float rbs[176];                    // rel bias for this head
  const int win  = blockIdx.x;
  const int head = blockIdx.y;
  const int lane = threadIdx.x & 31;
  const int wave = threadIdx.x >> 5;
  const size_t qrow = (size_t)win * NTOK;
  const int wloc = win & 63;
  const int wr7 = (wloc >> 3) * WINSZ, wc7 = (wloc & 7) * WINSZ;

  // ---- stage per-head rel-bias table
  if (threadIdx.x < 169) rbs[threadIdx.x] = relb[threadIdx.x * HEADS + head];

  // ---- stage Q, K, V (rows >= 49 zero-padded); V transposed for B operand
  {
    const int row = threadIdx.x >> 2;
    const int ch  = (threadIdx.x & 3) * 8;
    if (row < NTOK) {
      const bf16* src = qkv + (qrow + row) * 1152;
      *(uint4*)&Qs[row * 32 + ch] = *(const uint4*)&src[head * 32 + ch];
      *(uint4*)&Ks[row * 32 + ch] = *(const uint4*)&src[384 + head * 32 + ch];
      union { uint4 q; bf16 e[8]; } vt;
      vt.q = *(const uint4*)&src[768 + head * 32 + ch];
      #pragma unroll
      for (int j = 0; j < 8; ++j) Vt[(ch + j) * 64 + row] = vt.e[j];
    } else {
      uint4 z = {0u, 0u, 0u, 0u};
      *(uint4*)&Qs[row * 32 + ch] = z;
      *(uint4*)&Ks[row * 32 + ch] = z;
      #pragma unroll
      for (int j = 0; j < 8; ++j) Vt[(ch + j) * 64 + row] = (bf16)0.f;
    }
  }
  __syncthreads();

  // ---- build full additive table: rel-bias + mask; padded entries -> -1e30
  #pragma unroll
  for (int e = threadIdx.x; e < 64 * 64; e += 256) {
    const int n = e >> 6, m = e & 63;
    float v = -1e30f;
    if (n < NTOK && m < NTOK) {
      const int r1 = n / WINSZ, c1 = n - r1 * WINSZ;
      const int r2 = m / WINSZ, c2 = m - r2 * WINSZ;
      v = rbs[(r1 - r2 + 6) * 13 + (c1 - c2 + 6)];
      const int l1 = rgn(wr7 + r1) * 3 + rgn(wc7 + c1);
      const int l2 = rgn(wr7 + r2) * 3 + rgn(wc7 + c2);
      if (l1 != l2) v -= 10000.f;
    }
    addT[e] = v;
  }
  __syncthreads();

  // ---- S = Q K^T + addT (branchless epilogue)
  {
    const uint32_t* Q32 = (const uint32_t*)Qs;
    const uint32_t* K32 = (const uint32_t*)Ks;
    const int ti = wave >> 1;
    const int rowHi = (lane & 16) ? 8 : 0;
    #pragma unroll
    for (int jj = 0; jj < 2; ++jj) {
      const int tj = (wave & 1) * 2 + jj;
      floatx8 acc = {};
      bf16x16 af = fragA(Q32, ti * 16 + (lane & 15), 16, 0, lane);
      bf16x16 bfv = fragB(K32, tj * 16 + (lane & 15), 16, 0, lane);
      acc = wmma_bf16(af, bfv, acc);
      const int m = tj * 16 + (lane & 15);  // key token
      #pragma unroll
      for (int j = 0; j < 8; ++j) {
        const int n = ti * 16 + rowHi + j;  // query token
        Ss[n * 64 + m] = acc[j] + addT[n * 64 + m];
      }
    }
  }
  __syncthreads();

  // ---- softmax (4 lanes per row, 16 cols each)
  {
    const int r  = threadIdx.x >> 2;
    const int cq = (threadIdx.x & 3) * 16;
    float vr[16];
    float mx = -1e30f;
    #pragma unroll
    for (int j = 0; j < 16; ++j) { vr[j] = Ss[r * 64 + cq + j]; mx = fmaxf(mx, vr[j]); }
    mx = fmaxf(mx, __shfl_xor(mx, 1, 32));
    mx = fmaxf(mx, __shfl_xor(mx, 2, 32));
    float sum = 0.f;
    #pragma unroll
    for (int j = 0; j < 16; ++j) { vr[j] = __expf(vr[j] - mx); sum += vr[j]; }
    sum += __shfl_xor(sum, 1, 32);
    sum += __shfl_xor(sum, 2, 32);
    const float inv = 1.f / sum;
    #pragma unroll
    for (int j = 0; j < 16; ++j) Ps[r * 64 + cq + j] = (bf16)(vr[j] * inv);
  }
  __syncthreads();

  // ---- O = P V  (64x32 out = 8 tiles, one per wave; K=64 -> two WMMAs)
  {
    const uint32_t* P32 = (const uint32_t*)Ps;
    const uint32_t* V32 = (const uint32_t*)Vt;
    const int oti = wave >> 1;  // n-tile 0..3
    const int otj = wave & 1;   // d-tile 0..1
    floatx8 acc = {};
    #pragma unroll
    for (int kk = 0; kk < 64; kk += 32) {
      bf16x16 af = fragA(P32, oti * 16 + (lane & 15), 32, kk / 2, lane);
      bf16x16 bfv = fragB(V32, otj * 16 + (lane & 15), 32, kk / 2, lane);
      acc = wmma_bf16(af, bfv, acc);
    }
    const int col = head * 32 + otj * 16 + (lane & 15);
    const int rowHi = (lane & 16) ? 8 : 0;
    #pragma unroll
    for (int j = 0; j < 8; ++j) {
      const int n = oti * 16 + rowHi + j;
      if (n < NTOK) outp[(qrow + n) * DIMC + col] = (bf16)acc[j];
    }
  }
}

// ---------------------------------------------------------------------------
// Host launcher
// ---------------------------------------------------------------------------
extern "C" void kernel_launch(void* const* d_in, const int* in_sizes, int n_in,
                              void* d_out, int out_size, void* d_ws,
                              size_t ws_size, hipStream_t stream) {
  const float* x      = (const float*)d_in[0];
  const float* qkv_w  = (const float*)d_in[1];
  const float* qkv_b  = (const float*)d_in[2];
  const float* proj_w = (const float*)d_in[3];
  const float* proj_b = (const float*)d_in[4];
  const float* rel_b  = (const float*)d_in[5];
  const float* n1w    = (const float*)d_in[6];
  const float* n1b    = (const float*)d_in[7];
  const float* n2w    = (const float*)d_in[8];
  const float* n2b    = (const float*)d_in[9];
  const float* m1w    = (const float*)d_in[10];
  const float* m1b    = (const float*)d_in[11];
  const float* m2w    = (const float*)d_in[12];
  const float* m2b    = (const float*)d_in[13];

  char* ws = (char*)d_ws;
  size_t off = 0;
  auto take = [&](size_t bytes) -> char* {
    char* p = ws + off;
    off = (off + bytes + 255) & ~(size_t)255;
    return p;
  };
  bf16*  wt_qkv  = (bf16*)take((size_t)1152 * 384 * 2);
  bf16*  wt_proj = (bf16*)take((size_t)384 * 384 * 2);
  bf16*  wt_m1   = (bf16*)take((size_t)1536 * 384 * 2);
  bf16*  wt_m2   = (bf16*)take((size_t)384 * 1536 * 2);
  bf16*  xw      = (bf16*)take((size_t)MROWS * 384 * 2);   // LN1 out, reused for LN2 out
  bf16*  attno   = (bf16*)take((size_t)MROWS * 384 * 2);   // attention output
  float* xmid    = (float*)take((size_t)MROWS * 384 * 4);  // x after first residual
  bf16*  qkvh    = (bf16*)take((size_t)MROWS * 1536 * 2);  // qkv, reused for MLP hidden

  // weights -> bf16, transposed to (N x K)
  k_wconv<<<(384 * 1152 + 255) / 256, 256, 0, stream>>>(qkv_w, wt_qkv, 384, 1152);
  k_wconv<<<(384 * 384 + 255) / 256, 256, 0, stream>>>(proj_w, wt_proj, 384, 384);
  k_wconv<<<(384 * 1536 + 255) / 256, 256, 0, stream>>>(m1w, wt_m1, 384, 1536);
  k_wconv<<<(1536 * 384 + 255) / 256, 256, 0, stream>>>(m2w, wt_m2, 1536, 384);

  // LN1 + roll(-3,-3) + window partition
  k_ln<true><<<MROWS / 8, 256, 0, stream>>>(x, n1w, n1b, xw);

  // QKV projection (q pre-scaled)
  k_gemm<MODE_QKV><<<dim3(1152 / 64, MROWS / 128), 256, 0, stream>>>(
      xw, wt_qkv, qkv_b, qkvh, nullptr, 1152, 384);

  // windowed attention
  k_attn<<<dim3(BWIN, HEADS), 256, 0, stream>>>(qkvh, rel_b, attno);

  // proj + window-reverse + roll(+3,+3) + residual -> xmid (fp32)
  k_gemm<MODE_PROJ><<<dim3(384 / 64, MROWS / 128), 256, 0, stream>>>(
      attno, wt_proj, proj_b, xmid, x, 384, 384);

  // LN2
  k_ln<false><<<MROWS / 8, 256, 0, stream>>>(xmid, n2w, n2b, xw);

  // MLP up + exact GELU
  k_gemm<MODE_GELU><<<dim3(1536 / 64, MROWS / 128), 256, 0, stream>>>(
      xw, wt_m1, m1b, qkvh, nullptr, 1536, 384);

  // MLP down + residual -> d_out (fp32)
  k_gemm<MODE_MLP2><<<dim3(384 / 64, MROWS / 128), 256, 0, stream>>>(
      qkvh, wt_m2, m2b, d_out, xmid, 384, 1536);
}
